// DAPTriAttEnd_19112604467523
// MI455X (gfx1250) — matrix operand order repaired and checked by reading, not compiled, not run on hardware
//
#include <hip/hip_runtime.h>
#include <hip/hip_bf16.h>

typedef _Float16 v4h  __attribute__((ext_vector_type(4)));
typedef _Float16 v8h  __attribute__((ext_vector_type(8)));
typedef _Float16 v16h __attribute__((ext_vector_type(16)));
typedef float    v8f  __attribute__((ext_vector_type(8)));
typedef unsigned int u32x4 __attribute__((ext_vector_type(4)));
typedef int          i32x4 __attribute__((ext_vector_type(4)));
typedef int          i32x8 __attribute__((ext_vector_type(8)));

#define Nn 256
#define Cc 128
#define Hh 4
#define Dd 32

static __device__ inline v16h cat16(v8h lo, v8h hi) {
    return __builtin_shufflevector(lo, hi, 0,1,2,3,4,5,6,7,8,9,10,11,12,13,14,15);
}
static __device__ inline v8f wmma_f16(v16h a, v16h b, v8f c) {
    return __builtin_amdgcn_wmma_f32_16x16x32_f16(false, a, false, b, (short)0, c, false, false);
}
static __device__ inline v8f vzero8() { v8f z = {0.f,0.f,0.f,0.f,0.f,0.f,0.f,0.f}; return z; }

// ---------------------------------------------------------------------------
// Kernel 0: weight prep.  Wt[o][c] (o in [0,512): q|k|v|g concatenated) f16,
// with 1/sqrt(D) folded into q rows.  Wo16[o][c_in] f16 (already B^T layout).
// ---------------------------------------------------------------------------
__global__ __launch_bounds__(256) void prep_weights(
    const float* __restrict__ wq, const float* __restrict__ wk,
    const float* __restrict__ wv, const float* __restrict__ wg,
    const float* __restrict__ wo,
    _Float16* __restrict__ Wt, _Float16* __restrict__ Wo16)
{
    int idx = blockIdx.x * 256 + threadIdx.x;
    if (idx < 512 * Cc) {
        int o = idx >> 7, c = idx & 127;
        int blk = o >> 7, oo = o & 127;
        const float* src = (blk == 0) ? wq : (blk == 1) ? wk : (blk == 2) ? wv : wg;
        float val = src[c * 128 + oo];
        if (blk == 0) val *= 0.17677669529663687f;   // 1/sqrt(32)
        Wt[idx] = (_Float16)val;
    } else if (idx < 512 * Cc + Cc * Cc) {
        int j = idx - 512 * Cc;
        Wo16[j] = (_Float16)wo[j];                    // wo[o][k] row-major == B^T
    }
}

// ---------------------------------------------------------------------------
// Kernel 1: LayerNorm + fused QKVG projection (WMMA) + triangle bias.
// One block of 128 threads (4 waves) handles 16 consecutive positions
// p = i*N + j (same i).  Wave w owns output columns [128w,128w+128):
// wave0->q, wave1->k, wave2->v(transposed store), wave3->g.
// ---------------------------------------------------------------------------
__global__ __launch_bounds__(128) void ln_qkvg(
    const float* __restrict__ x,  const float* __restrict__ lnw,
    const float* __restrict__ lnb, const float* __restrict__ wbias,
    const _Float16* __restrict__ Wt,
    _Float16* __restrict__ qb, _Float16* __restrict__ kb,
    _Float16* __restrict__ vtb, _Float16* __restrict__ gb,
    float* __restrict__ tb)
{
    __shared__ __align__(16) _Float16 XN[16][136];   // padded, 16B-aligned rows

    const int tid = threadIdx.x;
    const int w = tid >> 5, l = tid & 31;
    const int p0 = blockIdx.x * 16;
    const int i  = p0 >> 8;          // constant within tile
    const int j0 = p0 & 255;

    // -------- LayerNorm: wave w handles rows 4w..4w+3, fp32 --------
    #pragma unroll
    for (int rr = 0; rr < 4; ++rr) {
        int r = w * 4 + rr;
        const float* row = x + ((size_t)(j0 + r) * Nn + i) * Cc;
        float4 xv = reinterpret_cast<const float4*>(row)[l];   // c = 4l..4l+3
        float s  = xv.x + xv.y + xv.z + xv.w;
        float s2 = xv.x*xv.x + xv.y*xv.y + xv.z*xv.z + xv.w*xv.w;
        #pragma unroll
        for (int m = 16; m >= 1; m >>= 1) {
            s  += __shfl_xor(s,  m, 32);
            s2 += __shfl_xor(s2, m, 32);
        }
        float mean = s * (1.0f / 128.0f);
        float var  = s2 * (1.0f / 128.0f) - mean * mean;
        float rstd = rsqrtf(var + 1e-5f);
        float4 wv4 = reinterpret_cast<const float4*>(lnw)[l];
        float4 bv4 = reinterpret_cast<const float4*>(lnb)[l];
        v4h o;
        o[0] = (_Float16)((xv.x - mean) * rstd * wv4.x + bv4.x);
        o[1] = (_Float16)((xv.y - mean) * rstd * wv4.y + bv4.y);
        o[2] = (_Float16)((xv.z - mean) * rstd * wv4.z + bv4.z);
        o[3] = (_Float16)((xv.w - mean) * rstd * wv4.w + bv4.w);
        *reinterpret_cast<v4h*>(&XN[r][4 * l]) = o;
    }
    __syncthreads();

    // -------- triangle bias: tb[h][i][j] = xn[i][j] . wbias[h]  --------
    if (tid < 64) {
        int r = tid >> 2, h = tid & 3;
        float acc = 0.f;
        #pragma unroll 4
        for (int c = 0; c < Cc; ++c)
            acc += (float)XN[r][c] * wbias[h * Cc + c];
        tb[((size_t)h * Nn + i) * Nn + (j0 + r)] = acc;
    }

    // -------- GEMM: 16 x 128 per wave via 8 WMMA tiles x 4 k-steps --------
    const int r  = l & 15;
    const int hi = l >> 4;
    v8f acc[8];
    #pragma unroll
    for (int t = 0; t < 8; ++t) acc[t] = vzero8();

    #pragma unroll
    for (int kk = 0; kk < 4; ++kk) {
        int k0 = 32 * kk;
        v8h alo = *reinterpret_cast<const v8h*>(&XN[r][k0 + hi * 16]);
        v8h ahi = *reinterpret_cast<const v8h*>(&XN[r][k0 + hi * 16 + 8]);
        v16h afrag = cat16(alo, ahi);
        #pragma unroll
        for (int t = 0; t < 8; ++t) {
            int o = w * 128 + t * 16 + r;                  // B^T row
            const _Float16* bp = Wt + (size_t)o * Cc + k0 + hi * 16;
            v16h bfrag = cat16(*reinterpret_cast<const v8h*>(bp),
                               *reinterpret_cast<const v8h*>(bp + 8));
            acc[t] = wmma_f16(afrag, bfrag, acc[t]);
        }
    }

    // -------- scatter stores (class uniform per wave) --------
    #pragma unroll
    for (int t = 0; t < 8; ++t) {
        int ol = t * 16 + r;                                // 0..127 within class
        #pragma unroll
        for (int v = 0; v < 8; ++v) {
            int m = v + 8 * hi;
            _Float16 val = (_Float16)acc[t][v];
            if (w == 0) {
                int h = ol >> 5, d = ol & 31;
                qb[((size_t)(i * Hh + h) * Nn + (j0 + m)) * Dd + d] = val;
            } else if (w == 1) {
                int h = ol >> 5, d = ol & 31;
                kb[((size_t)(i * Hh + h) * Nn + (j0 + m)) * Dd + d] = val;
            } else if (w == 2) {
                int h = ol >> 5, d = ol & 31;
                vtb[((size_t)(i * Hh + h) * Dd + d) * Nn + (j0 + m)] = val;
            } else {
                gb[(size_t)(p0 + m) * Cc + ol] = val;
            }
        }
    }
}

// ---------------------------------------------------------------------------
// Kernel 2: attention for one (i, h) pair per block (128 threads / 4 waves).
// K tile staged into LDS by the Tensor Data Mover (TENSOR_LOAD_TO_LDS,
// tracked by TENSORcnt); QK^T -> biases -> softmax -> PV via WMMA (K-dim=D=32).
// ---------------------------------------------------------------------------
__global__ __launch_bounds__(128) void attn(
    const _Float16* __restrict__ qb, const _Float16* __restrict__ kb,
    const _Float16* __restrict__ vtb, const float* __restrict__ tb,
    const float* __restrict__ mask, _Float16* __restrict__ ob)
{
    __shared__ __align__(16) _Float16 Ksh[Nn * Dd];       // 16 KB
    __shared__ __align__(16) _Float16 Psh[4][16 * 256];   // 32 KB (per-wave tiles)
    __shared__ float MB[Nn];                              // 1 KB

    const int tid = threadIdx.x;
    const int w = tid >> 5, l = tid & 31;
    const int i = blockIdx.x >> 2, h = blockIdx.x & 3;
    const int r = l & 15, hi = l >> 4;

    const size_t ih = (size_t)(i * Hh + h);
    const _Float16* ksrc = kb  + ih * Nn * Dd;
    const _Float16* vsrc = vtb + ih * Dd * Nn;

    // ---- TDM: one wave DMAs the whole 8192-element K tile into LDS ----
    if (w == 0) {
        unsigned lds_off = (unsigned)(uintptr_t)(void*)&Ksh[0];
        unsigned long long ga = (unsigned long long)(uintptr_t)ksrc;
        u32x4 g0;
        g0[0] = 1u;                                   // count=1, user descriptor
        g0[1] = lds_off;                              // lds_addr (bytes)
        g0[2] = (unsigned)(ga & 0xffffffffu);         // global_addr[31:0]
        g0[3] = (unsigned)((ga >> 32) & 0x1ffffffu)   // global_addr[56:32]
              | (2u << 30);                           // type = 2 ("image")
        i32x8 g1;
        g1[0] = 0x00010000;                           // data_size = 1 (2 bytes)
        g1[1] = (int)(8192u << 16);                   // tensor_dim0[15:0] = 8192
        g1[2] = 0x00010000;                           // dim0 hi=0 | tensor_dim1 = 1
        g1[3] = (int)(8192u << 16);                   // dim1 hi=0 | tile_dim0 = 8192
        g1[4] = 0;                                    // tile_dim1/2 unused
        g1[5] = 8192;                                 // tensor_dim0_stride lo
        g1[6] = 0; g1[7] = 0;
        i32x4 gz4 = {0, 0, 0, 0};                     // dims 2..4 unused
        i32x8 gz8 = {0, 0, 0, 0, 0, 0, 0, 0};
        __builtin_amdgcn_tensor_load_to_lds(g0, g1, gz4, gz4, gz8, 0);
        __builtin_amdgcn_s_wait_tensorcnt(0);
    }
    for (int u = tid; u < Nn; u += 128)
        MB[u] = 1e9f * (mask[(size_t)u * Nn + i] - 1.0f);
    __builtin_prefetch(tb + ((size_t)h * Nn) * Nn + tid * 64, 0, 1);
    __syncthreads();

    for (int qi = 0; qi < 4; ++qi) {
        const int j0 = (w * 4 + qi) * 16;

        // ---- Q A-fragment (global, contiguous) ----
        const _Float16* qsrc = qb + (ih * Nn + j0) * Dd;
        v16h qfrag = cat16(*reinterpret_cast<const v8h*>(qsrc + r * Dd + hi * 16),
                           *reinterpret_cast<const v8h*>(qsrc + r * Dd + hi * 16 + 8));

        // ---- S = Q K^T + mask_bias + tri_bias : 16 key-tiles ----
        v8f s[16];
        #pragma unroll
        for (int b = 0; b < 16; ++b) {
            const _Float16* kp = Ksh + (b * 16 + r) * Dd + hi * 16;
            v16h kfrag = cat16(*reinterpret_cast<const v8h*>(kp),
                               *reinterpret_cast<const v8h*>(kp + 8));
            v8f c = wmma_f16(qfrag, kfrag, vzero8());
            int key = b * 16 + r;                         // N-index of this lane
            float mb = MB[key];
            const float* tp = tb + ((size_t)h * Nn + (j0 + 8 * hi)) * Nn + key;
            #pragma unroll
            for (int v = 0; v < 8; ++v)
                c[v] += mb + tp[(size_t)v * Nn];
            s[b] = c;
        }

        // ---- softmax over the 256 keys of each row (16-lane N-groups) ----
        #pragma unroll
        for (int v = 0; v < 8; ++v) {
            float mx = -1e30f;
            #pragma unroll
            for (int b = 0; b < 16; ++b) mx = fmaxf(mx, s[b][v]);
            #pragma unroll
            for (int m = 8; m >= 1; m >>= 1) mx = fmaxf(mx, __shfl_xor(mx, m, 32));
            float sum = 0.f;
            #pragma unroll
            for (int b = 0; b < 16; ++b) {
                float e = __expf(s[b][v] - mx);
                s[b][v] = e; sum += e;
            }
            #pragma unroll
            for (int m = 8; m >= 1; m >>= 1) sum += __shfl_xor(sum, m, 32);
            float inv = __builtin_amdgcn_rcpf(sum);
            int mrow = v + 8 * hi;
            #pragma unroll
            for (int b = 0; b < 16; ++b)
                Psh[w][mrow * 256 + b * 16 + r] = (_Float16)(s[b][v] * inv);
        }
        __syncthreads();   // uniform; orders P stores before fragment reloads

        // ---- O = P V : 8 key-chunks of 32, two d-tiles ----
        v8f o0 = vzero8(), o1 = vzero8();
        #pragma unroll
        for (int c32 = 0; c32 < 8; ++c32) {
            const _Float16* pp = &Psh[w][r * 256 + c32 * 32];
            v16h pfrag = cat16(*reinterpret_cast<const v8h*>(pp + hi * 16),
                               *reinterpret_cast<const v8h*>(pp + hi * 16 + 8));
            const _Float16* v0p = vsrc + (size_t)r * Nn + c32 * 32 + hi * 16;
            const _Float16* v1p = vsrc + (size_t)(16 + r) * Nn + c32 * 32 + hi * 16;
            v16h b0 = cat16(*reinterpret_cast<const v8h*>(v0p),
                            *reinterpret_cast<const v8h*>(v0p + 8));
            v16h b1 = cat16(*reinterpret_cast<const v8h*>(v1p),
                            *reinterpret_cast<const v8h*>(v1p + 8));
            o0 = wmma_f16(pfrag, b0, o0);
            o1 = wmma_f16(pfrag, b1, o1);
        }

        #pragma unroll
        for (int v = 0; v < 8; ++v) {
            int m = v + 8 * hi;
            size_t base = ((size_t)i * Nn + (j0 + m)) * Cc + h * Dd;
            ob[base + r]      = (_Float16)o0[v];
            ob[base + 16 + r] = (_Float16)o1[v];
        }
    }
}

// ---------------------------------------------------------------------------
// Kernel 3: sigmoid gating + output projection (WMMA) + bias + swapaxes store.
// 128 threads / block, 16 positions per block; wave w owns out cols [32w,32w+32).
// ---------------------------------------------------------------------------
__global__ __launch_bounds__(128) void gate_out(
    const _Float16* __restrict__ ob, const _Float16* __restrict__ gb,
    const float* __restrict__ bg, const _Float16* __restrict__ Wo16,
    const float* __restrict__ bo, float* __restrict__ out)
{
    __shared__ __align__(16) _Float16 A[16][136];

    const int tid = threadIdx.x;
    const int w = tid >> 5, l = tid & 31;
    const int p0 = blockIdx.x * 16;
    const int i  = p0 >> 8;
    const int j0 = p0 & 255;

    // gating: thread -> (row r, 16-col segment)
    {
        int r = tid >> 3, c0 = (tid & 7) * 16;
        const _Float16* op = ob + (size_t)(p0 + r) * Cc + c0;
        const _Float16* gp = gb + (size_t)(p0 + r) * Cc + c0;
        v8h olo = *reinterpret_cast<const v8h*>(op);
        v8h ohi = *reinterpret_cast<const v8h*>(op + 8);
        v8h glo = *reinterpret_cast<const v8h*>(gp);
        v8h ghi = *reinterpret_cast<const v8h*>(gp + 8);
        #pragma unroll
        for (int e = 0; e < 8; ++e) {
            float gv  = (float)glo[e] + bg[c0 + e];
            float gv2 = (float)ghi[e] + bg[c0 + 8 + e];
            float s1 = __builtin_amdgcn_rcpf(1.0f + __expf(-gv));
            float s2 = __builtin_amdgcn_rcpf(1.0f + __expf(-gv2));
            A[r][c0 + e]     = (_Float16)((float)olo[e] * s1);
            A[r][c0 + 8 + e] = (_Float16)((float)ohi[e] * s2);
        }
    }
    __syncthreads();

    const int r = l & 15, hi = l >> 4;
    v8f acc0 = vzero8(), acc1 = vzero8();
    #pragma unroll
    for (int kk = 0; kk < 4; ++kk) {
        int k0 = 32 * kk;
        v16h afrag = cat16(*reinterpret_cast<const v8h*>(&A[r][k0 + hi * 16]),
                           *reinterpret_cast<const v8h*>(&A[r][k0 + hi * 16 + 8]));
        const _Float16* b0p = Wo16 + (size_t)(w * 32 + r) * Cc + k0 + hi * 16;
        const _Float16* b1p = Wo16 + (size_t)(w * 32 + 16 + r) * Cc + k0 + hi * 16;
        v16h b0 = cat16(*reinterpret_cast<const v8h*>(b0p),
                        *reinterpret_cast<const v8h*>(b0p + 8));
        v16h b1 = cat16(*reinterpret_cast<const v8h*>(b1p),
                        *reinterpret_cast<const v8h*>(b1p + 8));
        acc0 = wmma_f16(afrag, b0, acc0);
        acc1 = wmma_f16(afrag, b1, acc1);
    }

    int n0 = w * 32 + r, n1 = w * 32 + 16 + r;
    float bo0 = bo[n0], bo1 = bo[n1];
    #pragma unroll
    for (int v = 0; v < 8; ++v) {
        int m = v + 8 * hi;
        int j = j0 + m;
        size_t base = ((size_t)j * Nn + i) * Cc;   // swapaxes(1,2) on store
        out[base + n0] = acc0[v] + bo0;
        out[base + n1] = acc1[v] + bo1;
    }
}

// ---------------------------------------------------------------------------
extern "C" void kernel_launch(void* const* d_in, const int* in_sizes, int n_in,
                              void* d_out, int out_size, void* d_ws, size_t ws_size,
                              hipStream_t stream) {
    (void)in_sizes; (void)n_in; (void)out_size; (void)ws_size;
    const float* x     = (const float*)d_in[0];
    const float* mask  = (const float*)d_in[1];
    const float* ln_w  = (const float*)d_in[2];
    const float* ln_b  = (const float*)d_in[3];
    const float* wbias = (const float*)d_in[4];
    const float* wq    = (const float*)d_in[5];
    const float* wk    = (const float*)d_in[6];
    const float* wv    = (const float*)d_in[7];
    const float* wg    = (const float*)d_in[8];
    const float* bg    = (const float*)d_in[9];
    const float* wo    = (const float*)d_in[10];
    const float* bo    = (const float*)d_in[11];
    float* outp = (float*)d_out;

    char* ws = (char*)d_ws;
    _Float16* Wt   = (_Float16*)ws; ws += (size_t)512 * 128 * 2;
    _Float16* Wo16 = (_Float16*)ws; ws += (size_t)128 * 128 * 2;
    _Float16* qb   = (_Float16*)ws; ws += (size_t)Nn * Hh * Nn * Dd * 2;
    _Float16* kb   = (_Float16*)ws; ws += (size_t)Nn * Hh * Nn * Dd * 2;
    _Float16* vtb  = (_Float16*)ws; ws += (size_t)Nn * Hh * Dd * Nn * 2;
    _Float16* gb   = (_Float16*)ws; ws += (size_t)Nn * Nn * Cc * 2;
    _Float16* ob   = (_Float16*)ws; ws += (size_t)Nn * Nn * Cc * 2;
    float*    tb   = (float*)ws;    ws += (size_t)Hh * Nn * Nn * 4;

    prep_weights<<<(512 * 128 + 128 * 128 + 255) / 256, 256, 0, stream>>>(
        wq, wk, wv, wg, wo, Wt, Wo16);
    ln_qkvg<<<(Nn * Nn) / 16, 128, 0, stream>>>(
        x, ln_w, ln_b, wbias, Wt, qb, kb, vtb, gb, tb);
    attn<<<Nn * Hh, 128, 0, stream>>>(qb, kb, vtb, tb, mask, ob);
    gate_out<<<(Nn * Nn) / 16, 128, 0, stream>>>(ob, gb, bg, Wo16, bo, outp);
}
